// TempestBlock_80642305950367
// MI455X (gfx1250) — matrix operand order repaired and checked by reading, not compiled, not run on hardware
//
#include <hip/hip_runtime.h>
#include <hip/hip_bf16.h>
#include <math.h>

// ---------------- problem constants ----------------
constexpr int BB    = 2;
constexpr int TT    = 4096;
constexpr int DD    = 2048;
constexpr int DCONV = 1024;
constexpr int FFN   = 8192;
constexpr int M_    = BB * TT;          // 8192 rows
constexpr int N1    = 6144;             // conv_proj (3072) ++ aiv (3072)

typedef __attribute__((ext_vector_type(16))) __bf16 v16bf;
typedef __attribute__((ext_vector_type(8)))  float  v8f;
typedef unsigned short u16;

// ---------------- helpers ----------------
__device__ inline u16 f2bf(float x) {
  union { float f; unsigned u; } c; c.f = x;
  unsigned r = 0x7FFFu + ((c.u >> 16) & 1u);     // round-to-nearest-even
  return (u16)((c.u + r) >> 16);
}
__device__ inline float sigm(float x) { return 1.0f / (1.0f + __expf(-x)); }

// CDNA5 async copy: cache/memory -> LDS, no VGPR staging, tracked by ASYNCcnt.
// Generic pointers to LDS carry the LDS byte offset in their low 32 bits.
__device__ inline void async_ld128(void* lds, const void* gsrc) {
  asm volatile("global_load_async_to_lds_b128 %0, %1, off"
               :
               : "v"((unsigned)(size_t)lds),
                 "v"((unsigned long long)(size_t)gsrc)
               : "memory");
}
// Partial wait: allow N async ops to remain in flight (in-order completion
// means waiting to <=N retires everything older than the newest N ops).
template <int N>
__device__ inline void async_wait() {
  asm volatile("s_wait_asynccnt %0" :: "n"(N) : "memory");
}

union Frag16 { v16bf v; uint4 q[2]; };

// A-matrix 16x32 bf16 fragment (ISA 7.12.2), ld = row stride in u16:
// lanes 0-15: V0-3 K=0..7, V4-7 K=16..23 ; lanes 16-31: K=8..15 / K=24..31
__device__ inline v16bf load_a_frag(const u16* s, int ld, int row0, int lane) {
  const int r = row0 + (lane & 15);
  const int h = (lane >> 4) * 8;
  const u16* p = s + r * ld + h;
  Frag16 f;
  f.q[0] = *(const uint4*)(p);
  f.q[1] = *(const uint4*)(p + 16);
  return f.v;
}
// B-matrix 32x16 bf16 fragment: lane = column n; lanes 0-15 K=0..7,16..23 ; 16-31 K=8..15,24..31
__device__ inline v16bf load_b_frag(const u16* s, int ld, int col0, int lane) {
  const int c = col0 + (lane & 15);
  const int h = (lane >> 4) * 8;
  const u16* p = s + c * ld + h;
  Frag16 f;
  f.q[0] = *(const uint4*)(p);
  f.q[1] = *(const uint4*)(p + 16);
  return f.v;
}

__device__ inline float block_reduce_sum(float v) {
  #pragma unroll
  for (int o = 16; o > 0; o >>= 1) v += __shfl_xor(v, o, 32);
  __shared__ float red[8];
  const int lane = threadIdx.x & 31, w = threadIdx.x >> 5;
  if (lane == 0) red[w] = v;
  __syncthreads();
  float s = 0.f;
  #pragma unroll
  for (int i = 0; i < 8; ++i) s += red[i];
  return s;
}

// ---------------- fp32 -> bf16 conversion ----------------
__global__ void __launch_bounds__(256)
cvt_bf16_kernel(const float* __restrict__ in, u16* __restrict__ out, int n) {
  int i = blockIdx.x * blockDim.x + threadIdx.x;
  const int stride = gridDim.x * blockDim.x;
  for (; i < n; i += stride) out[i] = f2bf(in[i]);
}

// ---------------- rmsnorm -> bf16 ----------------
__global__ void __launch_bounds__(256)
rmsnorm_bf16_kernel(const float* __restrict__ x, const float* __restrict__ w,
                    u16* __restrict__ out) {
  const size_t base = (size_t)blockIdx.x * DD;
  float v[8]; float ss = 0.f;
  #pragma unroll
  for (int j = 0; j < 8; ++j) {
    v[j] = x[base + j * 256 + threadIdx.x];
    ss += v[j] * v[j];
  }
  const float inv = rsqrtf(block_reduce_sum(ss) * (1.0f / DD) + 1e-6f);
  #pragma unroll
  for (int j = 0; j < 8; ++j) {
    const int c = j * 256 + threadIdx.x;
    out[base + c] = f2bf(v[j] * inv * w[c]);
  }
}

// ---------------- WMMA GEMM: C[M,N] = A[M,K] * W[N,K]^T (+Add), bf16 in, f32 out
// 3-stage async LDS pipeline (ASYNCcnt partial waits), BK=64 -> 8 WMMA / barrier.
constexpr int BM = 128, BN = 64, BK = 64, DEPTH = 3;

template <int EPI>   // 0: C=acc ; 1: C=acc+Add
__global__ void __launch_bounds__(256)
gemm_bf16_kernel(const u16* __restrict__ A, const u16* __restrict__ W,
                 float* __restrict__ C, const float* __restrict__ Add,
                 int M, int N, int K) {
  __shared__ __align__(16) u16 As[DEPTH][BM * BK];   // 3 x 16 KB
  __shared__ __align__(16) u16 Bs[DEPTH][BN * BK];   // 3 x  8 KB
  const int tid  = threadIdx.x;
  const int lane = tid & 31;
  const int wave = tid >> 5;
  const int wr = (wave & 3) * 32;
  const int wc = (wave >> 2) * 32;
  const int gm = blockIdx.y * BM;
  const int gn = blockIdx.x * BN;

  const v8f vz = {0.f,0.f,0.f,0.f,0.f,0.f,0.f,0.f};
  v8f acc[2][2] = {{vz, vz}, {vz, vz}};

  constexpr int S = 6;                               // async ops per thread per stage
  const int arow = tid >> 1, acol = (tid & 1) * 32;  // 128x64 A tile: 64B/thread
  const int brow = tid >> 2, bcol = (tid & 3) * 16;  //  64x64 B tile: 32B/thread
  const u16* aptr = A + (size_t)(gm + arow) * K + acol;
  const u16* bptr = W + (size_t)(gn + brow) * K + bcol;

  auto stage = [&](int s, int kt) {
    const int k0 = kt * BK;
    #pragma unroll
    for (int j = 0; j < 4; ++j)
      async_ld128(&As[s][arow * BK + acol + j * 8], aptr + k0 + j * 8);
    #pragma unroll
    for (int j = 0; j < 2; ++j)
      async_ld128(&Bs[s][brow * BK + bcol + j * 8], bptr + k0 + j * 8);
  };
  auto compute = [&](int buf) {
    #pragma unroll
    for (int kk = 0; kk < 2; ++kk) {                 // two 32-K steps per tile
      const u16* ab = &As[buf][kk * 32];
      const u16* bb = &Bs[buf][kk * 32];
      v16bf a0 = load_a_frag(ab, BK, wr,      lane);
      v16bf a1 = load_a_frag(ab, BK, wr + 16, lane);
      v16bf b0 = load_b_frag(bb, BK, wc,      lane);
      v16bf b1 = load_b_frag(bb, BK, wc + 16, lane);
      acc[0][0] = __builtin_amdgcn_wmma_f32_16x16x32_bf16(false, a0, false, b0, (short)0, acc[0][0], false, false);
      acc[0][1] = __builtin_amdgcn_wmma_f32_16x16x32_bf16(false, a0, false, b1, (short)0, acc[0][1], false, false);
      acc[1][0] = __builtin_amdgcn_wmma_f32_16x16x32_bf16(false, a1, false, b0, (short)0, acc[1][0], false, false);
      acc[1][1] = __builtin_amdgcn_wmma_f32_16x16x32_bf16(false, a1, false, b1, (short)0, acc[1][1], false, false);
    }
  };

  const int nk = K / BK;                             // >= 32 for all our shapes
  stage(0, 0);
  stage(1, 1);
  async_wait<S>();                                   // stage 0 resident, stage 1 in flight
  __syncthreads();

  int ki = 0;
  for (; ki + 2 < nk; ++ki) {
    stage((ki + 2) % DEPTH, ki + 2);                 // overlaps with compute below
    compute(ki % DEPTH);
    async_wait<S>();                                 // retire stage ki+1 only
    __syncthreads();
  }
  compute(ki % DEPTH);                               // ki == nk-2
  async_wait<0>();                                   // last stage must land
  __syncthreads();
  ++ki;
  compute(ki % DEPTH);                               // ki == nk-1

  const int cn = lane & 15;
  const int mb = (lane >> 4) * 8;
  #pragma unroll
  for (int mf = 0; mf < 2; ++mf)
    #pragma unroll
    for (int nf = 0; nf < 2; ++nf) {
      const int m0 = gm + wr + mf * 16 + mb;
      const int n0 = gn + wc + nf * 16 + cn;
      #pragma unroll
      for (int r = 0; r < 8; ++r) {
        const size_t idx = (size_t)(m0 + r) * N + n0;
        float v = acc[mf][nf][r];
        if (EPI == 1) v += Add[idx];
        C[idx] = v;
      }
    }
}

// ---------------- fused gate/up GEMM with silu(g)*u epilogue -> bf16 ----------
__global__ void __launch_bounds__(256)
gemm_gateup_kernel(const u16* __restrict__ A, const u16* __restrict__ Wgu,
                   u16* __restrict__ Act, int M, int Nh, int K) {
  __shared__ __align__(16) u16 As[DEPTH][BM * BK];   // 3 x 16 KB
  __shared__ __align__(16) u16 Bg[DEPTH][BN * BK];   // 3 x  8 KB
  __shared__ __align__(16) u16 Bu[DEPTH][BN * BK];   // 3 x  8 KB
  const int tid  = threadIdx.x;
  const int lane = tid & 31;
  const int wave = tid >> 5;
  const int wr = (wave & 3) * 32;
  const int wc = (wave >> 2) * 32;
  const int gm = blockIdx.y * BM;
  const int gn = blockIdx.x * BN;

  const v8f vz = {0.f,0.f,0.f,0.f,0.f,0.f,0.f,0.f};
  v8f accg[2][2] = {{vz, vz}, {vz, vz}};
  v8f accu[2][2] = {{vz, vz}, {vz, vz}};

  constexpr int S = 8;
  const int arow = tid >> 1, acol = (tid & 1) * 32;
  const int brow = tid >> 2, bcol = (tid & 3) * 16;
  const u16* aptr = A + (size_t)(gm + arow) * K + acol;
  const u16* gptr = Wgu + (size_t)(gn + brow) * K + bcol;        // gate rows
  const u16* uptr = Wgu + (size_t)(Nh + gn + brow) * K + bcol;   // up rows

  auto stage = [&](int s, int kt) {
    const int k0 = kt * BK;
    #pragma unroll
    for (int j = 0; j < 4; ++j)
      async_ld128(&As[s][arow * BK + acol + j * 8], aptr + k0 + j * 8);
    #pragma unroll
    for (int j = 0; j < 2; ++j) {
      async_ld128(&Bg[s][brow * BK + bcol + j * 8], gptr + k0 + j * 8);
      async_ld128(&Bu[s][brow * BK + bcol + j * 8], uptr + k0 + j * 8);
    }
  };
  auto compute = [&](int buf) {
    #pragma unroll
    for (int kk = 0; kk < 2; ++kk) {
      const u16* ab = &As[buf][kk * 32];
      const u16* gb = &Bg[buf][kk * 32];
      const u16* ub = &Bu[buf][kk * 32];
      v16bf a0 = load_a_frag(ab, BK, wr,      lane);
      v16bf a1 = load_a_frag(ab, BK, wr + 16, lane);
      v16bf g0 = load_b_frag(gb, BK, wc,      lane);
      v16bf g1 = load_b_frag(gb, BK, wc + 16, lane);
      v16bf u0 = load_b_frag(ub, BK, wc,      lane);
      v16bf u1 = load_b_frag(ub, BK, wc + 16, lane);
      accg[0][0] = __builtin_amdgcn_wmma_f32_16x16x32_bf16(false, a0, false, g0, (short)0, accg[0][0], false, false);
      accg[0][1] = __builtin_amdgcn_wmma_f32_16x16x32_bf16(false, a0, false, g1, (short)0, accg[0][1], false, false);
      accg[1][0] = __builtin_amdgcn_wmma_f32_16x16x32_bf16(false, a1, false, g0, (short)0, accg[1][0], false, false);
      accg[1][1] = __builtin_amdgcn_wmma_f32_16x16x32_bf16(false, a1, false, g1, (short)0, accg[1][1], false, false);
      accu[0][0] = __builtin_amdgcn_wmma_f32_16x16x32_bf16(false, a0, false, u0, (short)0, accu[0][0], false, false);
      accu[0][1] = __builtin_amdgcn_wmma_f32_16x16x32_bf16(false, a0, false, u1, (short)0, accu[0][1], false, false);
      accu[1][0] = __builtin_amdgcn_wmma_f32_16x16x32_bf16(false, a1, false, u0, (short)0, accu[1][0], false, false);
      accu[1][1] = __builtin_amdgcn_wmma_f32_16x16x32_bf16(false, a1, false, u1, (short)0, accu[1][1], false, false);
    }
  };

  const int nk = K / BK;
  stage(0, 0);
  stage(1, 1);
  async_wait<S>();
  __syncthreads();

  int ki = 0;
  for (; ki + 2 < nk; ++ki) {
    stage((ki + 2) % DEPTH, ki + 2);
    compute(ki % DEPTH);
    async_wait<S>();
    __syncthreads();
  }
  compute(ki % DEPTH);
  async_wait<0>();
  __syncthreads();
  ++ki;
  compute(ki % DEPTH);

  const int cn = lane & 15;
  const int mb = (lane >> 4) * 8;
  #pragma unroll
  for (int mf = 0; mf < 2; ++mf)
    #pragma unroll
    for (int nf = 0; nf < 2; ++nf) {
      const int m0 = gm + wr + mf * 16 + mb;
      const int n0 = gn + wc + nf * 16 + cn;
      #pragma unroll
      for (int r = 0; r < 8; ++r) {
        const float g = accg[mf][nf][r];
        const float u = accu[mf][nf][r];
        Act[(size_t)(m0 + r) * Nh + n0] = f2bf(g * sigm(g) * u);
      }
    }
}

// ---------------- depthwise causal conv + Griffin scan (fused) ----------------
// P layout per row (6144 f32): [b(1024) | c(1024) | h(1024) | a(1024) | i(1024) | v(1024)]
__global__ void __launch_bounds__(256)
conv_scan_kernel(const float* __restrict__ P, const float* __restrict__ kern,
                 const float* __restrict__ cb, const float* __restrict__ db,
                 u16* __restrict__ mix) {
  const int tid = blockIdx.x * 256 + threadIdx.x;   // 0..2047
  const int b = tid >> 10, d = tid & 1023;
  const float k0 = kern[d * 3 + 0], k1 = kern[d * 3 + 1], k2 = kern[d * 3 + 2];
  const float cbias = cb[d], dbias = db[d];
  float y0 = 0.f, y1 = 0.f, hstate = 0.f;
  const float* Pr = P + (size_t)b * TT * (size_t)N1;
  u16* Mr = mix + (size_t)b * TT * (size_t)DD;
  for (int t = 0; t < TT; ++t) {
    const float* row = Pr + (size_t)t * N1;
    const float y2 = row[d] * row[2048 + d];                       // b*h
    const float conv = row[1024 + d] * (k0 * y0 + k1 * y1 + k2 * y2 + cbias);
    y0 = y1; y1 = y2;
    const float a  = sigm(row[3072 + d] + dbias);
    const float ig = sigm(row[4096 + d]);
    const float inp = sqrtf(fmaxf(1.0f - a * a, 1e-8f)) * (ig * row[5120 + d]);
    hstate = a * hstate + inp;                                      // == chunked_scan
    u16* mr = Mr + (size_t)t * DD;
    mr[d]        = f2bf(conv);
    mr[1024 + d] = f2bf(hstate);
  }
}

// -------- velocity EMA + residual + rmsnorm(h) -> bf16 (fused) --------
__global__ void __launch_bounds__(256)
residual_norm_kernel(const float* __restrict__ x, const float* __restrict__ vel,
                     const float* __restrict__ mo, const float* __restrict__ log_beta,
                     const float* __restrict__ ffn_w,
                     float* __restrict__ velout, float* __restrict__ h,
                     u16* __restrict__ hn) {
  const size_t base = (size_t)blockIdx.x * DD;
  const float beta = sigm(log_beta[0]);
  float hv[8]; float ss = 0.f;
  #pragma unroll
  for (int j = 0; j < 8; ++j) {
    const size_t i = base + j * 256 + threadIdx.x;
    const float vn = beta * vel[i] + mo[i];
    const float hh = x[i] + vn;
    velout[i] = vn;
    h[i] = hh;
    hv[j] = hh;
    ss += hh * hh;
  }
  const float inv = rsqrtf(block_reduce_sum(ss) * (1.0f / DD) + 1e-6f);
  #pragma unroll
  for (int j = 0; j < 8; ++j) {
    const int c = j * 256 + threadIdx.x;
    hn[base + c] = f2bf(hv[j] * inv * ffn_w[c]);
  }
}

// ---------------- launcher ----------------
extern "C" void kernel_launch(void* const* d_in, const int* in_sizes, int n_in,
                              void* d_out, int out_size, void* d_ws, size_t ws_size,
                              hipStream_t stream) {
  (void)in_sizes; (void)n_in; (void)out_size; (void)ws_size;
  const float* x           = (const float*)d_in[0];
  const float* vel         = (const float*)d_in[1];
  const float* pre_w       = (const float*)d_in[2];
  const float* conv_proj_w = (const float*)d_in[3];
  const float* conv_w      = (const float*)d_in[4];
  const float* conv_b      = (const float*)d_in[5];
  const float* w_aiv       = (const float*)d_in[6];
  const float* decay_bias  = (const float*)d_in[7];
  const float* out_proj_w  = (const float*)d_in[8];
  const float* log_beta    = (const float*)d_in[9];
  const float* ffn_w       = (const float*)d_in[10];
  const float* w_gate_up   = (const float*)d_in[11];
  const float* w_down      = (const float*)d_in[12];

  char* ws = (char*)d_ws;
  // ---- workspace layout (bytes) ----
  u16*   xn  = (u16*)  (ws);                                   //  33,554,432
  u16*   W1  = (u16*)  (ws + 33554432ull);                     //  25,165,824
  u16*   mix = (u16*)  (ws + 58720256ull);                     //  33,554,432
  u16*   Wo  = (u16*)  (ws + 92274688ull);                     //   8,388,608
  float* hbuf= (float*)(ws + 100663296ull);                    //  67,108,864
  u16*   hn  = (u16*)  (ws + 167772160ull);                    //  33,554,432
  u16*   Wgu = (u16*)  (ws + 201326592ull);                    //  67,108,864
  u16*   Wd  = (u16*)  (ws + 268435456ull);                    //  33,554,432
  float* Pf  = (float*)(ws + 301989888ull);                    // 201,326,592 (8192x6144 f32)
  float* mo  = Pf;                                             // reuse (P dead after conv_scan)
  u16*   act = (u16*)  (ws + 301989888ull + 67108864ull);      // reuse: 8192x8192 bf16

  float* out_f   = (float*)d_out;                // out     [8192 x 2048]
  float* vel_out = (float*)d_out + 16777216;     // vel_new [8192 x 2048]

  // 1) weight conversion (conv_proj || w_aiv fused into W1[6144,2048])
  cvt_bf16_kernel<<<2048, 256, 0, stream>>>(conv_proj_w, W1,            6291456);
  cvt_bf16_kernel<<<2048, 256, 0, stream>>>(w_aiv,       W1 + 6291456,  6291456);
  cvt_bf16_kernel<<<2048, 256, 0, stream>>>(out_proj_w,  Wo,            4194304);
  cvt_bf16_kernel<<<4096, 256, 0, stream>>>(w_gate_up,   Wgu,          33554432);
  cvt_bf16_kernel<<<2048, 256, 0, stream>>>(w_down,      Wd,           16777216);

  // 2) pre-norm -> bf16 activations
  rmsnorm_bf16_kernel<<<M_, 256, 0, stream>>>(x, pre_w, xn);

  // 3) fused projection GEMM: P = xn @ [conv_proj;w_aiv]^T   (8192x6144, K=2048)
  gemm_bf16_kernel<0><<<dim3(N1 / BN, M_ / BM), 256, 0, stream>>>(
      xn, W1, Pf, nullptr, M_, N1, DD);

  // 4) depthwise conv + Griffin scan -> bf16 mixer input [conv_out | scan_out]
  conv_scan_kernel<<<8, 256, 0, stream>>>(Pf, conv_w, conv_b, decay_bias, mix);

  // 5) mixer_out = mix @ out_proj^T (8192x2048, K=2048)
  gemm_bf16_kernel<0><<<dim3(DD / BN, M_ / BM), 256, 0, stream>>>(
      mix, Wo, mo, nullptr, M_, DD, DD);

  // 6) velocity EMA + residual + ffn rmsnorm
  residual_norm_kernel<<<M_, 256, 0, stream>>>(x, vel, mo, log_beta, ffn_w,
                                               vel_out, hbuf, hn);

  // 7) fused gate/up GEMM + silu epilogue -> act bf16 (8192x8192, K=2048)
  gemm_gateup_kernel<<<dim3(FFN / BN, M_ / BM), 256, 0, stream>>>(
      hn, Wgu, act, M_, FFN, DD);

  // 8) down GEMM + residual h -> d_out (8192x2048, K=8192)
  gemm_bf16_kernel<1><<<dim3(DD / BN, M_ / BM), 256, 0, stream>>>(
      act, Wd, out_f, hbuf, M_, DD, FFN);
}
